// NeuralNetworkTrainableMPO_7284264534629
// MI455X (gfx1250) — matrix operand order repaired and checked by reading, not compiled, not run on hardware
//
#include <hip/hip_runtime.h>
#include <hip/hip_bf16.h>
#include <math.h>

typedef __attribute__((ext_vector_type(16))) __bf16 v16bf;
typedef __attribute__((ext_vector_type(8)))  float  v8f;

#define B_DIM   8192
#define MPO_DIM 4096
#define HID_DIM 2048
#define OUT_DIM 1024
#define K_EMBED 1024   // L*2

// ---- workspace layout (bytes) ----
#define OFF_XBF   ((size_t)0)                     // 8192*1024 bf16 = 16 MB
#define OFF_WEBF  ((size_t)16777216)              // 4096*1024 bf16 =  8 MB
#define OFF_WHBF  ((size_t)25165824)              // 2048*4096 bf16 = 16 MB
#define OFF_WOBF  ((size_t)41943040)              // 1024*2048 bf16 =  4 MB
#define OFF_M     ((size_t)46137344)              // 16*4096 f32
#define OFF_S     ((size_t)46399488)              // 8192*16 f32
#define OFF_H2    ((size_t)46923776)              // 8192*4096 bf16 = 64 MB
#define OFF_H3    ((size_t)114032640)             // 8192*2048 bf16 = 32 MB

// ----------------------------------------------------------------------------
// fp32 -> bf16 convert (grid-stride)
// ----------------------------------------------------------------------------
__global__ void cvt_f32_bf16(const float* __restrict__ in, __bf16* __restrict__ out, size_t n) {
    size_t i = (size_t)blockIdx.x * blockDim.x + threadIdx.x;
    size_t stride = (size_t)gridDim.x * blockDim.x;
    for (; i < n; i += stride) out[i] = (__bf16)in[i];
}

__global__ void zero_f32(float* __restrict__ p, size_t n) {
    size_t i = (size_t)blockIdx.x * blockDim.x + threadIdx.x;
    size_t stride = (size_t)gridDim.x * blockDim.x;
    for (; i < n; i += stride) p[i] = 0.0f;
}

// ----------------------------------------------------------------------------
// Collapse MPO cores into M[16][4096]:
//   M[j, i*256+jj*16+j3] = sum_{l,t} core0[0][j,i,l] * cc1[l,jj,t] * cc2[t,j3]
// ----------------------------------------------------------------------------
__global__ void build_M_kernel(const float* __restrict__ core0,
                               const float* __restrict__ core1,
                               const float* __restrict__ core2,
                               float* __restrict__ M) {
    int pos = blockIdx.x * blockDim.x + threadIdx.x;   // 0..4095
    int i  = pos >> 8;
    int jj = (pos >> 4) & 15;
    int j3 = pos & 15;
    float cc1[2][2];
    #pragma unroll
    for (int l = 0; l < 2; ++l)
        #pragma unroll
        for (int t = 0; t < 2; ++t) {
            float sum = 0.f;
            for (int m = 0; m < 16; ++m)
                sum += core1[((l * 16 + jj) * 16 + m) * 2 + t];
            cc1[l][t] = sum * (1.0f / 16.0f);
        }
    float cc2[2];
    #pragma unroll
    for (int t = 0; t < 2; ++t) {
        float sum = 0.f;
        for (int m = 0; m < 16; ++m)
            sum += core2[(t * 16 + j3) * 16 + m];
        cc2[t] = sum * (1.0f / 16.0f);
    }
    for (int j = 0; j < 16; ++j) {
        float v = 0.f;
        #pragma unroll
        for (int l = 0; l < 2; ++l)
            #pragma unroll
            for (int t = 0; t < 2; ++t)
                v += core0[(j * 16 + i) * 2 + l] * cc1[l][t] * cc2[t];
        M[j * MPO_DIM + pos] = v;
    }
}

// ----------------------------------------------------------------------------
// h2[b,p] = tanh( s[b,:] @ M[:,p] + mpo_bias[p] )   (K=16 -> VALU)
// ----------------------------------------------------------------------------
__global__ void mpo_apply_kernel(const float* __restrict__ s,
                                 const float* __restrict__ M,
                                 const float* __restrict__ mpo_bias,
                                 __bf16* __restrict__ h2) {
    size_t idx = (size_t)blockIdx.x * blockDim.x + threadIdx.x;
    int p = (int)(idx & (MPO_DIM - 1));
    size_t b = idx >> 12;
    const float* sp = s + b * 16;
    float a = mpo_bias[p];
    #pragma unroll
    for (int j = 0; j < 16; ++j) a += sp[j] * M[j * MPO_DIM + p];
    h2[idx] = (__bf16)tanhf(a);
}

// ----------------------------------------------------------------------------
// WMMA bf16 GEMM:  out[row,col] = act( A[row,:] . W[col,:] + bias[col] )
//   A: (rows x K) bf16 row-major, W: (N x K) bf16 row-major  (A @ W^T)
//   block = 256 threads = 8 waves; block tile 256(M) x 64(N) x 32(K)
//   wave tile = 32(M) x 64(N): 8 WMMAs per K-step reusing 4 B fragments,
//   B staged through double-buffered LDS (one barrier per K-step).
//   MODE 0: tanh, reduce rows over 256-col groups into sred (no store)
//   MODE 1: tanh, store bf16
//   MODE 2: store f32
// ----------------------------------------------------------------------------
template<int MODE>
__global__ __launch_bounds__(256)
void wmma_gemm_kernel(const __bf16* __restrict__ A,
                      const __bf16* __restrict__ W,
                      const float*  __restrict__ bias,
                      void*         __restrict__ out,
                      float*        __restrict__ sred,
                      int K, int N) {
    __shared__ __bf16 Bs[2][64 * 32];  // 2 x (64 cols x 32 K) = 2 x 4 KB

    const int tid  = threadIdx.x;
    const int wave = tid >> 5;
    const int lane = tid & 31;
    const int l16  = lane & 15;
    const int hi   = lane >> 4;
    const int rowBase = blockIdx.x * 256 + wave * 32;
    const int colBase = blockIdx.y * 64;

    v8f acc[2][4];
    #pragma unroll
    for (int nt = 0; nt < 4; ++nt) {
        float bv = bias[colBase + nt * 16 + l16];
        #pragma unroll
        for (int e = 0; e < 8; ++e) { acc[0][nt][e] = bv; acc[1][nt][e] = bv; }
    }

    // A fragments: row = rowBase + rt*16 + l16; hi-half lanes own K=8..15 / 24..31
    const __bf16* aptr0 = A + (size_t)(rowBase + l16) * K + hi * 8;
    const __bf16* aptr1 = aptr0 + (size_t)16 * K;
    // B staging: each thread moves one 16B chunk per K step
    const int nl = tid >> 2;
    const int kc = (tid & 3) * 8;
    const __bf16* wptr = W + (size_t)(colBase + nl) * K + kc;

    // prologue: stage buffer 0
    *(uint4*)(&Bs[0][nl * 32 + kc]) = *(const uint4*)(wptr);
    __syncthreads();

    #pragma unroll 2
    for (int k0 = 0; k0 < K; k0 += 32) {
        const int buf = (k0 >> 5) & 1;
        if (k0 + 32 < K) {
            __builtin_prefetch(wptr + k0 + 64, 0, 1);       // next-next B tile
            *(uint4*)(&Bs[buf ^ 1][nl * 32 + kc]) = *(const uint4*)(wptr + k0 + 32);
        }

        union { v16bf v; uint4 q[2]; } a0, a1;
        a0.q[0] = *(const uint4*)(aptr0 + k0);
        a0.q[1] = *(const uint4*)(aptr0 + k0 + 16);
        a1.q[0] = *(const uint4*)(aptr1 + k0);
        a1.q[1] = *(const uint4*)(aptr1 + k0 + 16);

        #pragma unroll
        for (int nt = 0; nt < 4; ++nt) {
            union { v16bf v; uint4 q[2]; } bfr;
            const __bf16* bp = &Bs[buf][(nt * 16 + l16) * 32 + hi * 8];
            bfr.q[0] = *(const uint4*)(bp);
            bfr.q[1] = *(const uint4*)(bp + 16);
            acc[0][nt] = __builtin_amdgcn_wmma_f32_16x16x32_bf16(
                false, a0.v, false, bfr.v, (short)0, acc[0][nt], false, false);
            acc[1][nt] = __builtin_amdgcn_wmma_f32_16x16x32_bf16(
                false, a1.v, false, bfr.v, (short)0, acc[1][nt], false, false);
        }
        __syncthreads();
    }

    if (MODE == 0) {
        // tanh, sum rows across this 64-col tile, fold into s[row][colBase/256]
        float rs[2][8];
        #pragma unroll
        for (int rt = 0; rt < 2; ++rt)
            #pragma unroll
            for (int e = 0; e < 8; ++e) rs[rt][e] = 0.f;
        #pragma unroll
        for (int rt = 0; rt < 2; ++rt)
            #pragma unroll
            for (int nt = 0; nt < 4; ++nt)
                #pragma unroll
                for (int e = 0; e < 8; ++e) rs[rt][e] += tanhf(acc[rt][nt][e]);
        #pragma unroll
        for (int m = 1; m <= 8; m <<= 1)
            #pragma unroll
            for (int rt = 0; rt < 2; ++rt)
                #pragma unroll
                for (int e = 0; e < 8; ++e) rs[rt][e] += __shfl_xor(rs[rt][e], m, 32);
        if (l16 == 0) {
            const int grp = colBase >> 8;
            #pragma unroll
            for (int rt = 0; rt < 2; ++rt)
                #pragma unroll
                for (int e = 0; e < 8; ++e)
                    atomicAdd(&sred[(size_t)(rowBase + rt * 16 + hi * 8 + e) * 16 + grp],
                              rs[rt][e]);
        }
    } else {
        #pragma unroll
        for (int rt = 0; rt < 2; ++rt) {
            #pragma unroll
            for (int nt = 0; nt < 4; ++nt) {
                const int col = colBase + nt * 16 + l16;
                #pragma unroll
                for (int e = 0; e < 8; ++e) {
                    const int row = rowBase + rt * 16 + hi * 8 + e;
                    float v = acc[rt][nt][e];
                    if (MODE == 1) {
                        ((__bf16*)out)[(size_t)row * N + col] = (__bf16)tanhf(v);
                    } else {
                        ((float*)out)[(size_t)row * N + col] = v;
                    }
                }
            }
        }
    }
}

// ----------------------------------------------------------------------------
extern "C" void kernel_launch(void* const* d_in, const int* in_sizes, int n_in,
                              void* d_out, int out_size, void* d_ws, size_t ws_size,
                              hipStream_t stream) {
    const float* x        = (const float*)d_in[0];
    const float* W_embed  = (const float*)d_in[1];
    const float* b_embed  = (const float*)d_in[2];
    const float* core0    = (const float*)d_in[3];
    const float* core1    = (const float*)d_in[4];
    const float* core2    = (const float*)d_in[5];
    const float* mpo_bias = (const float*)d_in[6];
    const float* W_hidden = (const float*)d_in[7];
    const float* b_hidden = (const float*)d_in[8];
    const float* W_out    = (const float*)d_in[9];
    const float* b_out    = (const float*)d_in[10];

    char* ws = (char*)d_ws;
    __bf16* x_bf  = (__bf16*)(ws + OFF_XBF);
    __bf16* we_bf = (__bf16*)(ws + OFF_WEBF);
    __bf16* wh_bf = (__bf16*)(ws + OFF_WHBF);
    __bf16* wo_bf = (__bf16*)(ws + OFF_WOBF);
    float*  Mmat  = (float*)(ws + OFF_M);
    float*  sred  = (float*)(ws + OFF_S);
    __bf16* h2_bf = (__bf16*)(ws + OFF_H2);
    __bf16* h3_bf = (__bf16*)(ws + OFF_H3);

    // fp32 -> bf16 copies of activations/weights
    cvt_f32_bf16<<<4096, 256, 0, stream>>>(x,        x_bf,  (size_t)B_DIM * K_EMBED);
    cvt_f32_bf16<<<2048, 256, 0, stream>>>(W_embed,  we_bf, (size_t)MPO_DIM * K_EMBED);
    cvt_f32_bf16<<<4096, 256, 0, stream>>>(W_hidden, wh_bf, (size_t)HID_DIM * MPO_DIM);
    cvt_f32_bf16<<<1024, 256, 0, stream>>>(W_out,    wo_bf, (size_t)OUT_DIM * HID_DIM);

    zero_f32<<<512, 256, 0, stream>>>(sred, (size_t)B_DIM * 16);
    build_M_kernel<<<16, 256, 0, stream>>>(core0, core1, core2, Mmat);

    // Embed GEMM fused with tanh + group-sum reduction (h1 never materialized)
    wmma_gemm_kernel<0><<<dim3(B_DIM / 256, MPO_DIM / 64), 256, 0, stream>>>(
        x_bf, we_bf, b_embed, nullptr, sred, K_EMBED, MPO_DIM);

    // Collapsed MPO layer: h2 = tanh(s @ M + mpo_bias)
    mpo_apply_kernel<<<(B_DIM * MPO_DIM) / 256, 256, 0, stream>>>(sred, Mmat, mpo_bias, h2_bf);

    // Hidden GEMM: h3 = tanh(h2 @ W_hidden^T + b_hidden)
    wmma_gemm_kernel<1><<<dim3(B_DIM / 256, HID_DIM / 64), 256, 0, stream>>>(
        h2_bf, wh_bf, b_hidden, (void*)h3_bf, nullptr, MPO_DIM, HID_DIM);

    // Output GEMM: out = h3 @ W_out^T + b_out (fp32)
    wmma_gemm_kernel<2><<<dim3(B_DIM / 256, OUT_DIM / 64), 256, 0, stream>>>(
        h3_bf, wo_bf, b_out, d_out, nullptr, HID_DIM, OUT_DIM);
}
